// SS3D_20590073217680
// MI455X (gfx1250) — matrix operand order repaired and checked by reading, not compiled, not run on hardware
//
#include <hip/hip_runtime.h>
#include <hip/hip_bf16.h>

// SS3D (3D selective-scan / VMamba block) for gfx1250 (MI455X).
// [K1] in_proj GEMM  (WMMA f16->f32, 16x64 register block, A reused x4)
// [K2] depthwise 3x3x3 conv + GELU
// [K3] x_proj GEMM   (WMMA, gathered B fragment reused across 5 M-tiles)
// [K4] selective scan: block=(b,k), thread=d, h[64] in VGPRs,
//      x_dbl column staged in LDS via GLOBAL_LOAD_ASYNC_TO_LDS_B32 (ASYNCcnt)
// [K5] direction combine (deterministic)
// [K6] out_proj GEMM (WMMA, gathered A=(y*z) fragment reused x4)
// Workspace: ~250 MB in d_ws.

#define BATCH 32
#define LTOT  320      // 5*8*8
#define DMODEL 512
#define DINNER 512
#define KDIR  6
#define NSTATE 64
#define DTRANK 32
#define CTOT  160      // DTRANK + 2*NSTATE

typedef _Float16 h16;
typedef __attribute__((ext_vector_type(16))) _Float16 v16h;
typedef __attribute__((ext_vector_type(8)))  float    v8f;

__device__ __forceinline__ float gelu_exact(float x) {
    return 0.5f * x * (1.0f + erff(x * 0.7071067811865476f));
}
__device__ __forceinline__ float softplus_f(float x) {
    return (x > 20.0f) ? x : log1pf(expf(x));
}
// Scan position lp of direction k -> canonical (d,h,w) flat index.
// Serves both as gather (build xs) and scatter (un-permute out_y).
__device__ __forceinline__ int src_index(int k, int lp) {
    int l2 = (k < 3) ? lp : (LTOT - 1 - lp);
    int kk = (k < 3) ? k : k - 3;
    if (kk == 0) return l2;
    if (kk == 1) { int h = l2 / 40; int r = l2 - h * 40; int d = r >> 3; int w = r & 7;
                   return d * 64 + h * 8 + w; }
    int w = l2 / 40; int r = l2 - w * 40; int h = r / 5; int d = r - h * 5;
    return d * 64 + h * 8 + w;
}

// A fragment (16x32 f16, M x K), K contiguous in memory at pa[0..31].
__device__ __forceinline__ v16h load_afrag(const float* pa, int kg) {
    v16h a;
    #pragma unroll
    for (int i = 0; i < 8; ++i) a[i]     = (h16)pa[16 * kg + i];
    #pragma unroll
    for (int i = 0; i < 8; ++i) a[8 + i] = (h16)pa[16 * kg + 8 + i];
    return a;
}
// B fragment (32x16 f16, K x N), K contiguous at pb[0..31] for this lane's N.
__device__ __forceinline__ v16h load_bfrag(const float* pb, int kg) {
    v16h bf;
    #pragma unroll
    for (int i = 0; i < 16; ++i) bf[i] = (h16)pb[16 * kg + i];
    return bf;
}

// ---------------- K1: in_proj GEMM: M=B*L=10240, N=1024, K=512 --------------
// out[m,o] = sum_c x[m,c]*W[o,c]; wave does 16x64, A fragment reused x4.
__global__ __launch_bounds__(128)
void k_inproj(const float* __restrict__ x, const float* __restrict__ w,
              float* __restrict__ xin, float* __restrict__ zb)
{
    const int lane = threadIdx.x & 31;
    const int warp = threadIdx.x >> 5;
    int job = blockIdx.x * 4 + warp;          // 640 * 16 = 10240 jobs
    int mt = job >> 4, ns = job & 15;
    int m0 = mt * 16, n0 = ns * 64;
    int rn = lane & 15, kg = lane >> 4;

    const float* arow = x + (size_t)(m0 + rn) * DMODEL;
    const float* brow[4];
    #pragma unroll
    for (int j = 0; j < 4; ++j) brow[j] = w + (size_t)(n0 + 16 * j + rn) * DMODEL;

    v8f acc[4] = {};
    for (int kb = 0; kb < DMODEL; kb += 32) {
        __builtin_prefetch(arow + kb + 64, 0, 0);
        v16h a = load_afrag(arow + kb, kg);
        #pragma unroll
        for (int j = 0; j < 4; ++j) {
            v16h bf = load_bfrag(brow[j] + kb, kg);
            acc[j] = __builtin_amdgcn_wmma_f32_16x16x32_f16(
                false, a, false, bf, (short)0, acc[j], false, false);
        }
    }
    #pragma unroll
    for (int j = 0; j < 4; ++j) {
        int o = n0 + 16 * j + rn;
        #pragma unroll
        for (int r = 0; r < 8; ++r) {
            int m = m0 + r + 8 * kg;
            int b = m / LTOT, l = m - b * LTOT;
            float v = acc[j][r];
            if (o < DINNER)
                xin[((size_t)b * DINNER + o) * LTOT + l] = v;
            else
                zb[((size_t)b * DINNER + (o - DINNER)) * LTOT + l] = gelu_exact(v);
        }
    }
}

// ---------------- K2: depthwise conv3d 3x3x3 (pad 1) + GELU -----------------
__global__ void k_conv_gelu(const float* __restrict__ xin,
                            const float* __restrict__ cw,
                            float* __restrict__ xc)
{
    size_t idx = (size_t)blockIdx.x * blockDim.x + threadIdx.x;
    const size_t N1 = (size_t)BATCH * DINNER * LTOT;
    if (idx >= N1) return;
    int l  = (int)(idx % LTOT);
    size_t t = idx / LTOT;
    int d  = (int)(t % DINNER);
    int ww_ = l & 7, hh = (l >> 3) & 7, dd = l >> 6;
    const float* wp   = cw + (size_t)d * 27;
    const float* base = xin + (idx - l);
    float s = 0.f;
    #pragma unroll
    for (int kd = -1; kd <= 1; ++kd) {
        int zd = dd + kd; if (zd < 0 || zd >= 5) continue;
        #pragma unroll
        for (int kh = -1; kh <= 1; ++kh) {
            int zh = hh + kh; if (zh < 0 || zh >= 8) continue;
            #pragma unroll
            for (int kw2 = -1; kw2 <= 1; ++kw2) {
                int zw = ww_ + kw2; if (zw < 0 || zw >= 8) continue;
                s = fmaf(base[zd * 64 + zh * 8 + zw],
                         wp[(kd + 1) * 9 + (kh + 1) * 3 + (kw2 + 1)], s);
            }
        }
    }
    xc[idx] = gelu_exact(s);
}

// ---------------- K3: x_proj GEMM per (b,k): M=160(c), N=320(l), K=512(d) ---
// x_dbl[b,k,c,l] = sum_d W[k,c,d]*xc[b,d,src_index(k,l)]
// Gathered B fragment (16 strided loads) reused across 5 M-tiles (80 rows).
__global__ __launch_bounds__(128)
void k_xproj(const float* __restrict__ xpw, const float* __restrict__ xc,
             float* __restrict__ dbl)
{
    const int lane = threadIdx.x & 31;
    const int warp = threadIdx.x >> 5;
    int job = blockIdx.x * 4 + warp;          // 192 * 40 = 7680 jobs
    int bk = job / 40; int t = job - bk * 40;
    int ms = t / 20, nt = t - ms * 20;
    int b = bk / KDIR, k = bk - b * KDIR;
    int mbase = ms * 80, n0 = nt * 16;
    int rn = lane & 15, kg = lane >> 4;

    const float* arow[5];
    #pragma unroll
    for (int j = 0; j < 5; ++j)
        arow[j] = xpw + ((size_t)k * CTOT + mbase + 16 * j + rn) * DINNER;
    int sidx = src_index(k, n0 + rn);
    const float* bcol = xc + (size_t)b * DINNER * LTOT + sidx;

    v8f acc[5] = {};
    for (int kb = 0; kb < DINNER; kb += 32) {
        const float* pb = bcol + (size_t)(kb + 16 * kg) * LTOT;
        v16h bf;
        #pragma unroll
        for (int i = 0; i < 16; ++i) bf[i] = (h16)pb[(size_t)i * LTOT];
        #pragma unroll
        for (int j = 0; j < 5; ++j) {
            v16h a = load_afrag(arow[j] + kb, kg);
            acc[j] = __builtin_amdgcn_wmma_f32_16x16x32_f16(
                false, a, false, bf, (short)0, acc[j], false, false);
        }
    }
    #pragma unroll
    for (int j = 0; j < 5; ++j) {
        #pragma unroll
        for (int r = 0; r < 8; ++r) {
            int c = mbase + 16 * j + r + 8 * kg;
            int l = n0 + rn;
            dbl[((size_t)(b * KDIR + k) * CTOT + c) * LTOT + l] = acc[j][r];
        }
    }
}

// ---------------- K4: selective scan; block=(b,k), thread=d -----------------
__global__ __launch_bounds__(512)
void k_scan(const float* __restrict__ xc, const float* __restrict__ dbl,
            const float* __restrict__ dtw, const float* __restrict__ alogs,
            const float* __restrict__ Ds, float* __restrict__ ybuf)
{
    __shared__ float sh[CTOT];   // [0,32)=dt_low  [32,96)=B_t  [96,160)=C_t
    int b = blockIdx.x / KDIR, k = blockIdx.x % KDIR;
    int d = threadIdx.x;

    float Wdt[DTRANK];
    #pragma unroll
    for (int r = 0; r < DTRANK; ++r)
        Wdt[r] = dtw[((size_t)k * DINNER + d) * DTRANK + r];
    float An[NSTATE];
    #pragma unroll
    for (int n = 0; n < NSTATE; ++n)
        An[n] = -expf(alogs[((size_t)k * DINNER + d) * NSTATE + n]);
    float Dskip = Ds[k * DINNER + d];
    float h[NSTATE];
    #pragma unroll
    for (int n = 0; n < NSTATE; ++n) h[n] = 0.f;

    const float* ucol = xc  + (size_t)(b * DINNER + d) * LTOT;
    const float* dcol = dbl + (size_t)(b * KDIR + k) * CTOT * LTOT;
    float* ycol = ybuf + (size_t)((b * KDIR + k) * DINNER + d) * LTOT;

    // LDS byte address of sh[tid]: aperture places the LDS offset in addr[31:0]
    unsigned lds_addr = (unsigned)(uintptr_t)(&sh[0]) + (unsigned)threadIdx.x * 4u;

    for (int lp = 0; lp < LTOT; ++lp) {
        if (threadIdx.x < CTOT) {
            unsigned voff = ((unsigned)threadIdx.x * LTOT + (unsigned)lp) * 4u;
            // memory -> LDS, no VGPR round trip; tracked by ASYNCcnt
            asm volatile("global_load_async_to_lds_b32 %0, %1, %2"
                         :: "v"(lds_addr), "v"(voff), "s"(dcol) : "memory");
        }
        asm volatile("s_wait_asynccnt 0x0" ::: "memory");
        __syncthreads();

        int sidx = src_index(k, lp);
        float u = ucol[sidx];
        float dtr = 0.f;
        #pragma unroll
        for (int r = 0; r < DTRANK; ++r) dtr = fmaf(Wdt[r], sh[r], dtr);
        float dt = softplus_f(dtr);
        float du = dt * u;
        float y = 0.f;
        #pragma unroll
        for (int n = 0; n < NSTATE; ++n) {
            float dA = __expf(dt * An[n]);
            h[n] = fmaf(dA, h[n], du * sh[32 + n]);
            y = fmaf(h[n], sh[96 + n], y);
        }
        y = fmaf(u, Dskip, y);
        ycol[sidx] = y;          // bijective per k -> deterministic
        __syncthreads();
    }
}

// ---------------- K5: combine 6 directions (deterministic sum) --------------
__global__ void k_combine(const float* __restrict__ ybuf, float* __restrict__ ysum)
{
    size_t idx = (size_t)blockIdx.x * blockDim.x + threadIdx.x;
    const size_t N1 = (size_t)BATCH * DINNER * LTOT;
    if (idx >= N1) return;
    int l = (int)(idx % LTOT);
    size_t t = idx / LTOT;
    int d = (int)(t % DINNER);
    int b = (int)(t / DINNER);
    float s = 0.f;
    #pragma unroll
    for (int k = 0; k < KDIR; ++k)
        s += ybuf[(size_t)((b * KDIR + k) * DINNER + d) * LTOT + l];
    ysum[idx] = s;
}

// ---------------- K6: out_proj GEMM per b: M=320(l), N=512(o), K=512(c) -----
// out[b,l,o] = sum_c (ysum*z)[b,c,l]*Wo[o,c]; gathered A reused across 4 N-tiles.
__global__ __launch_bounds__(128)
void k_outproj(const float* __restrict__ ysum, const float* __restrict__ zb,
               const float* __restrict__ wo, float* __restrict__ out)
{
    const int lane = threadIdx.x & 31;
    const int warp = threadIdx.x >> 5;
    int job = blockIdx.x * 4 + warp;          // 32 * 160 = 5120 jobs
    int b = job / 160; int t = job - b * 160;
    int mt = t >> 3, ns = t & 7;
    int m0 = mt * 16, n0 = ns * 64;
    int rn = lane & 15, kg = lane >> 4;

    const float* acolY = ysum + (size_t)b * DINNER * LTOT + (m0 + rn);
    const float* acolZ = zb   + (size_t)b * DINNER * LTOT + (m0 + rn);
    const float* brow[4];
    #pragma unroll
    for (int j = 0; j < 4; ++j) brow[j] = wo + (size_t)(n0 + 16 * j + rn) * DINNER;

    v8f acc[4] = {};
    for (int kb = 0; kb < DINNER; kb += 32) {
        v16h a;
        #pragma unroll
        for (int i = 0; i < 8; ++i) {
            size_t k0 = (size_t)(kb + 16 * kg + i) * LTOT;
            size_t k1 = (size_t)(kb + 16 * kg + 8 + i) * LTOT;
            a[i]     = (h16)(acolY[k0] * acolZ[k0]);
            a[8 + i] = (h16)(acolY[k1] * acolZ[k1]);
        }
        #pragma unroll
        for (int j = 0; j < 4; ++j) {
            v16h bf = load_bfrag(brow[j] + kb, kg);
            acc[j] = __builtin_amdgcn_wmma_f32_16x16x32_f16(
                false, a, false, bf, (short)0, acc[j], false, false);
        }
    }
    #pragma unroll
    for (int j = 0; j < 4; ++j) {
        #pragma unroll
        for (int r = 0; r < 8; ++r) {
            int l = m0 + r + 8 * kg;
            int o = n0 + 16 * j + rn;
            out[((size_t)b * LTOT + l) * DMODEL + o] = acc[j][r];
        }
    }
}

extern "C" void kernel_launch(void* const* d_in, const int* in_sizes, int n_in,
                              void* d_out, int out_size, void* d_ws, size_t ws_size,
                              hipStream_t stream) {
    (void)in_sizes; (void)n_in; (void)out_size; (void)ws_size;
    const float* x      = (const float*)d_in[0];
    const float* ipw    = (const float*)d_in[1];
    const float* convw  = (const float*)d_in[2];
    const float* xpw    = (const float*)d_in[3];
    const float* dtw    = (const float*)d_in[4];
    const float* alogs  = (const float*)d_in[5];
    const float* Ds     = (const float*)d_in[6];
    const float* wo     = (const float*)d_in[7];
    float* out = (float*)d_out;

    const size_t N1   = (size_t)BATCH * DINNER * LTOT;        //  5,242,880
    const size_t NDBL = (size_t)BATCH * KDIR * CTOT * LTOT;   //  9,830,400
    const size_t NY   = (size_t)KDIR * N1;                    // 31,457,280
    float* ws   = (float*)d_ws;                               // total ~250 MB
    float* xin  = ws;
    float* zb   = xin + N1;
    float* xc   = zb + N1;
    float* dbl  = xc + N1;
    float* ybuf = dbl + NDBL;
    float* ysum = ybuf + NY;

    const int ew = 256;
    k_inproj  <<<2560, 128, 0, stream>>>(x, ipw, xin, zb);
    k_conv_gelu<<<(int)((N1 + ew - 1) / ew), ew, 0, stream>>>(xin, convw, xc);
    k_xproj   <<<1920, 128, 0, stream>>>(xpw, xc, dbl);
    k_scan    <<<BATCH * KDIR, DINNER, 0, stream>>>(xc, dbl, dtw, alogs, Ds, ybuf);
    k_combine <<<(int)((N1 + ew - 1) / ew), ew, 0, stream>>>(ybuf, ysum);
    k_outproj <<<1280, 128, 0, stream>>>(ysum, zb, wo, out);
}
